// SparseAttention_19490561589457
// MI455X (gfx1250) — matrix operand order repaired and checked by reading, not compiled, not run on hardware
//
#include <hip/hip_runtime.h>
#include <hip/hip_bf16.h>

// Problem constants (from reference)
#define B_ 2
#define S_ 2048
#define D_ 1024
#define H_ 16
#define DK_ 64
#define WIN_ 128
#define STRIDE_ 128

typedef __bf16 bf16;
typedef __attribute__((ext_vector_type(16))) __bf16 v16bf;
typedef __attribute__((ext_vector_type(8)))  float  v8f;
typedef __attribute__((ext_vector_type(4)))  float  v4f;   // clang vector (NT-store legal)
typedef unsigned int v4u __attribute__((ext_vector_type(4)));
typedef int          v4i __attribute__((ext_vector_type(4)));
typedef int          v8i __attribute__((ext_vector_type(8)));

union U16B { v16bf v; uint4 u[2]; };

// Load a 16-element bf16 fragment from two 16-byte chunks (global or LDS).
static __device__ inline v16bf ld_frag(const bf16* p0, const bf16* p1) {
  U16B r;
  r.u[0] = *(const uint4*)p0;
  r.u[1] = *(const uint4*)p1;
  return r.v;
}

static __device__ inline v8f wmma_bf16(v16bf a, v16bf b, v8f c) {
  // D = A(16x32) * B(32x16) + C, fp32 accumulate
  return __builtin_amdgcn_wmma_f32_16x16x32_bf16(false, a, false, b, (short)0, c,
                                                 false, false);
}

// Generic pointer to a __shared__ object: low 32 bits are the LDS byte offset
// (ISA: LDS aperture address = addr[31:0]).
static __device__ inline unsigned lds_off(const void* p) {
  return (unsigned)(unsigned long long)(uintptr_t)p;
}

// ---------------------------------------------------------------------------
// TDM: issue a 2D tile load (tile_w x tile_h bf16 elements, row stride
// stride_elems) from global memory into LDS at lds_byte_off. Descriptor per
// CDNA5 ISA D# groups 0/1 (2D: groups 2/3 zero).
// Must be executed by exactly one wave (TDM ignores EXEC). Tracked by
// TENSORcnt.
// ---------------------------------------------------------------------------
static __device__ inline void tdm_load_2d(unsigned lds_byte_off, const bf16* gptr,
                                          unsigned tile_w, unsigned tile_h,
                                          unsigned long long stride_elems) {
  const unsigned long long ga = (unsigned long long)(uintptr_t)gptr;
  v4u g0;
  g0.x = 1u;                                                // count=1 (valid)
  g0.y = lds_byte_off;                                      // lds_addr
  g0.z = (unsigned)ga;                                      // global_addr[31:0]
  g0.w = (unsigned)((ga >> 32) & 0x01FFFFFFu) | (2u << 30); // addr[56:32]|type=2

  v8i g1;
  g1[0] = 0x00010000;                                   // data_size=1 -> 2 bytes
  g1[1] = (int)((tile_w & 0xFFFFu) << 16);              // tensor_dim0[15:0]
  g1[2] = (int)((tile_w >> 16) | ((tile_h & 0xFFFFu) << 16)); // td0 hi | td1 lo
  g1[3] = (int)((tile_h >> 16) | ((tile_w & 0xFFFFu) << 16)); // td1 hi | tile_dim0
  g1[4] = (int)(tile_h & 0xFFFFu);                      // tile_dim1 (tile_dim2=0)
  g1[5] = (int)(unsigned)(stride_elems & 0xFFFFFFFFu);  // tensor_dim0_stride lo
  g1[6] = (int)(unsigned)(stride_elems >> 32);          // stride[47:32]
  g1[7] = 0;

  v4i gz = {0, 0, 0, 0};
#if defined(__clang_major__) && (__clang_major__ >= 23)
  v8i gz8 = {0, 0, 0, 0, 0, 0, 0, 0};
  __builtin_amdgcn_tensor_load_to_lds(g0, g1, gz, gz, gz8, 0);
#else
  __builtin_amdgcn_tensor_load_to_lds(g0, g1, gz, gz, 0);
#endif
}

// ---------------------------------------------------------------------------
// fp32 -> bf16 cast (grid-stride)
// ---------------------------------------------------------------------------
__global__ void cast_f32_bf16(const float* __restrict__ in, bf16* __restrict__ out,
                              int n) {
  int i = blockIdx.x * blockDim.x + threadIdx.x;
  int stride = gridDim.x * blockDim.x;
  for (; i < n; i += stride) out[i] = (bf16)in[i];
}

// ---------------------------------------------------------------------------
// C[MxN] = A[MxK] @ W^T (W is [N,K], both bf16 k-contiguous) + bias
// TDM double-buffered: wave0 DMAs the 128x32 A/B tiles for step k+1 into LDS
// while all 8 waves compute step k out of LDS. One s_barrier per step.
// mode 0: fp32 row-major out; mode 1: bf16 [b,h,s,dk]; mode 2: bf16 [b,h,dk,s]
// ---------------------------------------------------------------------------
__global__ __launch_bounds__(256)
void gemm_qkvo(const bf16* __restrict__ A, const bf16* __restrict__ W,
               const float* __restrict__ bias, void* __restrict__ outp,
               int M, int N, int K, int mode) {
  __shared__ bf16 At[2][128 * 32];
  __shared__ bf16 Bt[2][128 * 32];

  const int lane = threadIdx.x & 31;
  const int wave = __builtin_amdgcn_readfirstlane(threadIdx.x >> 5); // scalar!
  const int half = lane >> 4;
  const int l16  = lane & 15;
  const int rowBase = blockIdx.y * 128;
  const int colBase = blockIdx.x * 128;
  const int waveM = (wave & 3) * 32;
  const int waveN = (wave >> 2) * 64;

  v8f acc[2][4];
#pragma unroll
  for (int mt = 0; mt < 2; mt++)
#pragma unroll
    for (int nt = 0; nt < 4; nt++) acc[mt][nt] = (v8f){};

  const int ksteps = K / 32;
  if (wave == 0) {  // scalar branch: TDM ignores EXEC, must not be lane-masked
    tdm_load_2d(lds_off(&At[0][0]), A + (size_t)rowBase * K, 32, 128, K);
    tdm_load_2d(lds_off(&Bt[0][0]), W + (size_t)colBase * K, 32, 128, K);
  }

  for (int kk = 0; kk < ksteps; kk++) {
    const int cur = kk & 1;
    if (wave == 0) __builtin_amdgcn_s_wait_tensorcnt(0);
    __syncthreads();  // cur ready for all; all done reading buffer 'nxt'
    if (wave == 0 && (kk + 1) < ksteps) {
      const int nxt = cur ^ 1;
      tdm_load_2d(lds_off(&At[nxt][0]),
                  A + (size_t)rowBase * K + (size_t)(kk + 1) * 32, 32, 128, K);
      tdm_load_2d(lds_off(&Bt[nxt][0]),
                  W + (size_t)colBase * K + (size_t)(kk + 1) * 32, 32, 128, K);
    }

    v16bf afr[2], bfr[4];
#pragma unroll
    for (int mt = 0; mt < 2; mt++) {
      const bf16* pa = &At[cur][(waveM + mt * 16 + l16) * 32 + half * 16];
      afr[mt] = ld_frag(pa, pa + 8);
    }
#pragma unroll
    for (int nt = 0; nt < 4; nt++) {
      const bf16* pb = &Bt[cur][(waveN + nt * 16 + l16) * 32 + half * 16];
      bfr[nt] = ld_frag(pb, pb + 8);
    }
#pragma unroll
    for (int mt = 0; mt < 2; mt++)
#pragma unroll
      for (int nt = 0; nt < 4; nt++)
        acc[mt][nt] = wmma_bf16(afr[mt], bfr[nt], acc[mt][nt]);
  }

#pragma unroll
  for (int mt = 0; mt < 2; mt++) {
#pragma unroll
    for (int nt = 0; nt < 4; nt++) {
      const int col = colBase + waveN + nt * 16 + l16;
      const float bv = bias[col];
#pragma unroll
      for (int v = 0; v < 8; v++) {
        const int row = rowBase + waveM + mt * 16 + v + half * 8;
        const float val = acc[mt][nt][v] + bv;
        if (mode == 0) {
          ((float*)outp)[(size_t)row * N + col] = val;
        } else {
          const int b = row >> 11, s = row & (S_ - 1);
          const int h = col >> 6,  dk = col & (DK_ - 1);
          if (mode == 1)
            ((bf16*)outp)[(((size_t)(b * H_ + h)) * S_ + s) * DK_ + dk] = (bf16)val;
          else
            ((bf16*)outp)[(((size_t)(b * H_ + h)) * DK_ + dk) * S_ + s] = (bf16)val;
        }
      }
    }
  }
}

// ---------------------------------------------------------------------------
// Sparse attention. One wave per 16-query tile; 8 waves per block.
// Window key-tiles [t-8, t+8] (17) + 1 gathered strided tile (j%128==0).
// Scores in 18 x v8f registers; exact softmax; dense attn_weights rows are
// written with non-temporal stores (536 MB stream >> 192 MB L2).
// O = P@V via WMMA with P transposed to A-layout through a 1KB/wave LDS stage.
// ---------------------------------------------------------------------------
__global__ __launch_bounds__(256)
void sparse_attn(const bf16* __restrict__ Qb, const bf16* __restrict__ Kb,
                 const bf16* __restrict__ Vt, bf16* __restrict__ Ob,
                 float* __restrict__ attnOut) {
  __shared__ bf16 pstage[8][16 * 32];

  const int lane = threadIdx.x & 31;
  const int wave = threadIdx.x >> 5;
  const int half = lane >> 4;
  const int l16  = lane & 15;

  const int gw = blockIdx.x * 8 + wave;   // 4096 waves total
  const int bh = gw >> 7;                 // (b*H + h)
  const int t  = gw & 127;                // query tile index
  const int q0 = t * 16;

  const bf16* Qh = Qb + (size_t)bh * S_ * DK_;
  const bf16* Kh = Kb + (size_t)bh * S_ * DK_;
  const bf16* Vh = Vt + (size_t)bh * DK_ * S_;
  float* arow = attnOut + (size_t)bh * S_ * S_ + (size_t)q0 * S_;

  // ---- zero-fill the 16 x 2048 attn-weight rows this wave owns (NT stores)
  {
    const v4f z4 = (v4f){};
    for (int r = 0; r < 16; r++) {
      float* rp = arow + (size_t)r * S_;
      for (int c = lane * 4; c < S_; c += 128)
        __builtin_nontemporal_store(z4, (v4f*)(rp + c));
    }
  }

  // ---- Q A-fragments (16 x 64 = two 16x32 frags)
  const bf16* pq = Qh + (size_t)(q0 + l16) * DK_ + half * 16;
  const v16bf qa0 = ld_frag(pq,      pq + 8);
  const v16bf qa1 = ld_frag(pq + 32, pq + 40);

  v8f Sacc[18];
  const float scale = 0.125f;  // 1/sqrt(64)

  // ---- window score tiles (clamped addresses; masked later)
#pragma unroll
  for (int i = 0; i < 17; i++) {
    const int kt  = t - 8 + i;
    const int ktc = min(max(kt, 0), 127);
    const bf16* pk = Kh + (size_t)(ktc * 16 + l16) * DK_ + half * 16;
    const v16bf kb0 = ld_frag(pk,      pk + 8);    // dk 0..31
    const v16bf kb1 = ld_frag(pk + 32, pk + 40);   // dk 32..63
    v8f a = (v8f){};
    a = wmma_bf16(qa0, kb0, a);
    a = wmma_bf16(qa1, kb1, a);
    Sacc[i] = a;
  }
  // ---- strided score tile (column n -> key 128*n)
  {
    const bf16* pk = Kh + (size_t)(l16 * STRIDE_) * DK_ + half * 16;
    const v16bf kb0 = ld_frag(pk,      pk + 8);
    const v16bf kb1 = ld_frag(pk + 32, pk + 40);
    v8f a = (v8f){};
    a = wmma_bf16(qa0, kb0, a);
    a = wmma_bf16(qa1, kb1, a);
    Sacc[17] = a;
  }

  // ---- scale + mask (strided columns excluded from window tiles)
#pragma unroll
  for (int i = 0; i < 17; i++) {
    const int kt = t - 8 + i;
    const bool ktv = (kt >= 0) && (kt < 128);
    const int k = kt * 16 + l16;
#pragma unroll
    for (int v = 0; v < 8; v++) {
      const int q = q0 + v + half * 8;
      const int ad = (q > k) ? (q - k) : (k - q);
      const bool allowed = ktv && (ad <= WIN_) && ((k & (STRIDE_ - 1)) != 0);
      Sacc[i][v] = allowed ? Sacc[i][v] * scale : -__builtin_inff();
    }
  }
#pragma unroll
  for (int v = 0; v < 8; v++) Sacc[17][v] *= scale;

  // ---- exact softmax (row = v + half*8; columns live across 16 lanes)
#pragma unroll
  for (int v = 0; v < 8; v++) {
    float m = Sacc[17][v];
#pragma unroll
    for (int i = 0; i < 17; i++) m = fmaxf(m, Sacc[i][v]);
    m = fmaxf(m, __shfl_xor(m, 1));
    m = fmaxf(m, __shfl_xor(m, 2));
    m = fmaxf(m, __shfl_xor(m, 4));
    m = fmaxf(m, __shfl_xor(m, 8));

    float s = 0.f;
#pragma unroll
    for (int i = 0; i < 18; i++) {
      const float p = __expf(Sacc[i][v] - m);
      Sacc[i][v] = p;
      s += p;
    }
    s += __shfl_xor(s, 1);
    s += __shfl_xor(s, 2);
    s += __shfl_xor(s, 4);
    s += __shfl_xor(s, 8);
    const float inv = 1.f / s;
#pragma unroll
    for (int i = 0; i < 18; i++) Sacc[i][v] *= inv;
  }

  // ---- write attn weights (window tiles write 0 at strided cols; strided
  //      tile overwrites them last — same-wave stores are ordered)
#pragma unroll
  for (int i = 0; i < 17; i++) {
    const int kt = t - 8 + i;
    if (kt >= 0 && kt < 128) {
      const int k = kt * 16 + l16;
#pragma unroll
      for (int v = 0; v < 8; v++) {
        const int r = v + half * 8;
        __builtin_nontemporal_store(Sacc[i][v], &arow[(size_t)r * S_ + k]);
      }
    }
  }
#pragma unroll
  for (int v = 0; v < 8; v++) {
    const int r = v + half * 8;
    __builtin_nontemporal_store(Sacc[17][v], &arow[(size_t)r * S_ + l16 * STRIDE_]);
  }

  // ---- O = P @ V : process tile pairs; transpose P via LDS into A-layout
  v8f Oacc[4];
#pragma unroll
  for (int nt = 0; nt < 4; nt++) Oacc[nt] = (v8f){};

  bf16* myst = pstage[wave];

#pragma unroll
  for (int pi = 0; pi < 9; pi++) {
    const int i0 = pi * 2;
    // stage P[16x32] (cols 0-15 = tile i0, cols 16-31 = tile i0+1) as bf16
#pragma unroll
    for (int v = 0; v < 8; v++) {
      const int r = v + half * 8;
      myst[r * 32 + l16]      = (bf16)Sacc[i0][v];
      myst[r * 32 + 16 + l16] = (bf16)Sacc[i0 + 1][v];
    }
    // A-frag from LDS (wave-private; compiler inserts s_wait_dscnt)
    const bf16* pl = myst + l16 * 32 + half * 16;
    U16B pa_u;
    pa_u.u[0] = *(const uint4*)(pl);
    pa_u.u[1] = *(const uint4*)(pl + 8);
    const v16bf pa = pa_u.v;

    if (pi < 8) {
      int ks = (t - 8 + i0) * 16;
      ks = min(max(ks, 0), S_ - 32);   // clamped; invalid tiles have p == 0
#pragma unroll
      for (int nt = 0; nt < 4; nt++) {
        const bf16* pv = Vh + (size_t)(nt * 16 + l16) * S_ + ks + half * 16;
        const v16bf vb = ld_frag(pv, pv + 8);
        Oacc[nt] = wmma_bf16(pa, vb, Oacc[nt]);
      }
    } else {
      // pair (tile t+8, strided tile): lanes<16 contiguous, lanes>=16 gather
      const int ks = min((t + 8) * 16, S_ - 16);
#pragma unroll
      for (int nt = 0; nt < 4; nt++) {
        const int dk = nt * 16 + l16;
        v16bf vb;
        if (half == 0) {
          const bf16* pv = Vh + (size_t)dk * S_ + ks;
          vb = ld_frag(pv, pv + 8);
        } else {
          v16bf g;
#pragma unroll
          for (int c = 0; c < 16; c++) g[c] = Vh[(size_t)dk * S_ + c * STRIDE_];
          vb = g;
        }
        Oacc[nt] = wmma_bf16(pa, vb, Oacc[nt]);
      }
    }
  }

  // ---- store O (token-major bf16 [b, s, h*64+dk]) for the output GEMM
  const int b = bh >> 4, h = bh & 15;
#pragma unroll
  for (int nt = 0; nt < 4; nt++) {
#pragma unroll
    for (int v = 0; v < 8; v++) {
      const int s  = q0 + v + half * 8;
      const int dk = nt * 16 + l16;
      Ob[((size_t)(b * S_) + s) * D_ + h * DK_ + dk] = (bf16)Oacc[nt][v];
    }
  }
}

// ---------------------------------------------------------------------------
extern "C" void kernel_launch(void* const* d_in, const int* in_sizes, int n_in,
                              void* d_out, int out_size, void* d_ws, size_t ws_size,
                              hipStream_t stream) {
  (void)in_sizes; (void)n_in; (void)out_size; (void)ws_size;
  const float* x  = (const float*)d_in[0];
  const float* Wq = (const float*)d_in[1];
  const float* bq = (const float*)d_in[2];
  const float* Wk = (const float*)d_in[3];
  const float* bk = (const float*)d_in[4];
  const float* Wv = (const float*)d_in[5];
  const float* bv = (const float*)d_in[6];
  const float* Wo = (const float*)d_in[7];
  const float* bo = (const float*)d_in[8];

  char* ws = (char*)d_ws;
  bf16* xb  = (bf16*)(ws);                          //  8 MiB  x bf16
  bf16* Wqb = (bf16*)(ws + ((size_t)8  << 20));     //  2 MiB each
  bf16* Wkb = (bf16*)(ws + ((size_t)10 << 20));
  bf16* Wvb = (bf16*)(ws + ((size_t)12 << 20));
  bf16* Wob = (bf16*)(ws + ((size_t)14 << 20));
  bf16* Qb  = (bf16*)(ws + ((size_t)16 << 20));     //  8 MiB [b,h,s,dk]
  bf16* Kb  = (bf16*)(ws + ((size_t)24 << 20));     //  8 MiB [b,h,s,dk]
  bf16* Vt  = (bf16*)(ws + ((size_t)32 << 20));     //  8 MiB [b,h,dk,s]
  bf16* Ob  = (bf16*)(ws + ((size_t)40 << 20));     //  8 MiB [b,s,d]

  float* out0 = (float*)d_out;
  float* attn = out0 + (size_t)B_ * S_ * D_;

  cast_f32_bf16<<<256, 256, 0, stream>>>(x,  xb,  B_ * S_ * D_);
  cast_f32_bf16<<<64,  256, 0, stream>>>(Wq, Wqb, D_ * D_);
  cast_f32_bf16<<<64,  256, 0, stream>>>(Wk, Wkb, D_ * D_);
  cast_f32_bf16<<<64,  256, 0, stream>>>(Wv, Wvb, D_ * D_);
  cast_f32_bf16<<<64,  256, 0, stream>>>(Wo, Wob, D_ * D_);

  dim3 g(D_ / 128, (B_ * S_) / 128);  // 8 x 32
  gemm_qkvo<<<g, 256, 0, stream>>>(xb, Wqb, bq, Qb, B_ * S_, D_, D_, 1);
  gemm_qkvo<<<g, 256, 0, stream>>>(xb, Wkb, bk, Kb, B_ * S_, D_, D_, 1);
  gemm_qkvo<<<g, 256, 0, stream>>>(xb, Wvb, bv, Vt, B_ * S_, D_, D_, 2);

  sparse_attn<<<512, 256, 0, stream>>>(Qb, Kb, Vt, Ob, attn);

  gemm_qkvo<<<g, 256, 0, stream>>>(Ob, Wob, bo, out0, B_ * S_, D_, D_, 0);
}
